// EGNN_11751030522774
// MI455X (gfx1250) — matrix operand order repaired and checked by reading, not compiled, not run on hardware
//
#include <hip/hip_runtime.h>
#include <hip/hip_bf16.h>

typedef __attribute__((ext_vector_type(16))) _Float16 v16h;
typedef __attribute__((ext_vector_type(8)))  _Float16 v8h;
typedef __attribute__((ext_vector_type(8)))  float    v8f;

#define T_STEPS 10
#define HDIM    64
#define NNODES  5000
#define NEDGES  80000
#define NLAYERS 4
#define FOUT    8

__device__ __forceinline__ float silu_f(float x) {
    return x * (1.0f / (1.0f + __expf(-x)));
}

__device__ __forceinline__ v8f wmma_f16(v16h a, v16h b, v8f c) {
    // D = A(16x32 f16) * B(32x16 f16) + C(16x16 f32)
    return __builtin_amdgcn_wmma_f32_16x16x32_f16(false, a, false, b, (short)0, c,
                                                  false, false);
}

// A-fragment (16x32 f16) for this lane from an LDS tile staged row-major with
// `stride` f16 per row. ISA layout: lanes 0-15 hold M=lane, K runs {0..7,16..23};
// lanes 16-31 hold M=lane-16, K runs {8..15,24..31} (relative to kbase).
__device__ __forceinline__ v16h load_afrag(const _Float16* buf, int stride,
                                           int lane, int kbase) {
    int m = lane & 15;
    int half = lane >> 4;
    union { v16h v; v8h h[2]; } a;
    a.h[0] = *(const v8h*)(buf + m * stride + kbase + half * 8);
    a.h[1] = *(const v8h*)(buf + m * stride + kbase + 16 + half * 8);
    return a.v;
}

// ---------------------------------------------------------------------------
// Weight packing: row-major f32 [K,64] -> B-fragment order f16:
//   pack[kstep][ntile][lane][j], element = W[kstep*32 + (j/8)*16 + (lane/16)*8 + j%8][ntile*16 + lane%16]
// so GEMM-side loads are one contiguous 32B v16h per lane.
// ---------------------------------------------------------------------------
__global__ void pack_kernel(const float* __restrict__ src, _Float16* __restrict__ dst,
                            int K) {
    int p = blockIdx.x * 256 + threadIdx.x;
    if (p >= K * 64) return;
    int j    = p & 15;
    int lane = (p >> 4) & 31;
    int nt   = (p >> 9) & 3;
    int kk   = p >> 11;
    int k = kk * 32 + (j >> 3) * 16 + (lane >> 4) * 8 + (j & 7);
    int n = nt * 16 + (lane & 15);
    dst[p] = (_Float16)src[k * 64 + n];
}

// node embedding: h32/hh[(n*T+t)*64+k] = h[n]*emb_w[k] + emb_b[k] + time_emb[t][k]
__global__ void embed_kernel(const float* __restrict__ h, const float* __restrict__ emb_w,
                             const float* __restrict__ emb_b, const float* __restrict__ time_emb,
                             float* __restrict__ h32, _Float16* __restrict__ hh, int total) {
    int idx = blockIdx.x * 256 + threadIdx.x;
    if (idx >= total) return;
    int k = idx & 63;
    int r = idx >> 6;
    int n = r / T_STEPS;
    int t = r - n * T_STEPS;
    float v = h[n] * emb_w[k] + emb_b[k] + time_emb[t * HDIM + k];
    h32[idx] = v;
    hh[idx] = (_Float16)v;
}

// coord[(n*T+t)*3+c] = x[t][n][c]
__global__ void coordinit_kernel(const float* __restrict__ x, float* __restrict__ coord,
                                 int total) {
    int idx = blockIdx.x * 256 + threadIdx.x;
    if (idx >= total) return;
    int c = idx % 3;
    int r = idx / 3;
    int n = r / T_STEPS;
    int t = r - n * T_STEPS;
    coord[idx] = x[((size_t)t * NNODES + n) * 3 + c];
}

__global__ void count_kernel(const int* __restrict__ rowIdx, int* __restrict__ cnt, int E) {
    int i = blockIdx.x * 256 + threadIdx.x;
    if (i < E) atomicAdd(&cnt[rowIdx[i]], 1);
}

// ---------------------------------------------------------------------------
// Edge kernel: one wave32 per 16-row tile of the (E*T) edge-time rows
// (r = e*T + t).  Fused: radial -> edge MLP (2 layers) -> coord MLP,
// scattering m into magg and coord_diff*cm into agg via f32 atomics.
// GEMM1 is split:  out = h[row]@W1a + h[col]@W1b + rank-3(radial, eattr).
// ---------------------------------------------------------------------------
__global__ __launch_bounds__(256) void edge_kernel(
    const _Float16* __restrict__ hh, const float* __restrict__ coord,
    const int* __restrict__ rowIdx, const int* __restrict__ colIdx,
    const float* __restrict__ eattr,
    const _Float16* __restrict__ W1a, const _Float16* __restrict__ W1b,
    const _Float16* __restrict__ W2p, const _Float16* __restrict__ Cw1p,
    const float* __restrict__ b1, const float* __restrict__ w1c,
    const float* __restrict__ wd0, const float* __restrict__ wd1,
    const float* __restrict__ b2, const float* __restrict__ cb1,
    const float* __restrict__ cw2,
    float* __restrict__ magg, float* __restrict__ agg, int numTiles) {
    __shared__ __align__(16) _Float16 s_h[8][16 * HDIM]; // per-wave staging
    __shared__ float s_rad[8][16];
    __shared__ float s_cd[8][16][3];
    __shared__ float s_ea[8][16][2];
    __shared__ int   s_dst[8][16];
    __shared__ float s_cm[8][16];

    int w = threadIdx.x >> 5;
    int lane = threadIdx.x & 31;
    int tile = blockIdx.x * 8 + w;
    if (tile >= numTiles) return;
    int tb = tile * 16;

    // per-row metadata (lanes 0..15)
    if (lane < 16) {
        int r = tb + lane;
        int e = r / T_STEPS;
        int t = r - e * T_STEPS;
        int rn = rowIdx[e], cn = colIdx[e];
        const float* cr = coord + ((size_t)rn * T_STEPS + t) * 3;
        const float* cc = coord + ((size_t)cn * T_STEPS + t) * 3;
        float d0 = cr[0] - cc[0], d1 = cr[1] - cc[1], d2 = cr[2] - cc[2];
        s_cd[w][lane][0] = d0; s_cd[w][lane][1] = d1; s_cd[w][lane][2] = d2;
        s_rad[w][lane] = d0 * d0 + d1 * d1 + d2 * d2;
        s_ea[w][lane][0] = eattr[e * 2 + 0];
        s_ea[w][lane][1] = eattr[e * 2 + 1];
        s_dst[w][lane] = rn * T_STEPS + t;
        s_cm[w][lane] = 0.0f;
    }

    const v8f vzero = {0.f, 0.f, 0.f, 0.f, 0.f, 0.f, 0.f, 0.f};
    v8f acc[4];
#pragma unroll
    for (int nt = 0; nt < 4; ++nt) acc[nt] = vzero;

    // gather h[row] (f16, 64B per lane) into staging
    {
        int i = lane >> 1, part = lane & 1;
        int r = tb + i;
        int e = r / T_STEPS;
        int t = r - e * T_STEPS;
        int n = rowIdx[e];
        const uint4* src = (const uint4*)(hh + ((size_t)n * T_STEPS + t) * HDIM + part * 32);
        uint4* dst = (uint4*)(&s_h[w][i * HDIM + part * 32]);
        dst[0] = src[0]; dst[1] = src[1]; dst[2] = src[2]; dst[3] = src[3];
    }
    const v16h* Ba = (const v16h*)W1a;
#pragma unroll
    for (int kk = 0; kk < 2; ++kk) {
        v16h a = load_afrag(&s_h[w][0], HDIM, lane, kk * 32);
#pragma unroll
        for (int nt = 0; nt < 4; ++nt)
            acc[nt] = wmma_f16(a, Ba[(kk * 4 + nt) * 32 + lane], acc[nt]);
    }

    // gather h[col] (DS ops are in-order per wave: prior frag loads complete first)
    {
        int i = lane >> 1, part = lane & 1;
        int r = tb + i;
        int e = r / T_STEPS;
        int t = r - e * T_STEPS;
        int n = colIdx[e];
        const uint4* src = (const uint4*)(hh + ((size_t)n * T_STEPS + t) * HDIM + part * 32);
        uint4* dst = (uint4*)(&s_h[w][i * HDIM + part * 32]);
        dst[0] = src[0]; dst[1] = src[1]; dst[2] = src[2]; dst[3] = src[3];
    }
    const v16h* Bb = (const v16h*)W1b;
#pragma unroll
    for (int kk = 0; kk < 2; ++kk) {
        v16h a = load_afrag(&s_h[w][0], HDIM, lane, kk * 32);
#pragma unroll
        for (int nt = 0; nt < 4; ++nt)
            acc[nt] = wmma_f16(a, Bb[(kk * 4 + nt) * 32 + lane], acc[nt]);
    }

    int mbase = (lane >> 4) * 8;
    int nl = lane & 15;

    // epilogue 1: + bias + rank-3(radial, eattr), SiLU, restage as f16
#pragma unroll
    for (int nt = 0; nt < 4; ++nt) {
        int n = nt * 16 + nl;
        float bias = b1[n], c1 = w1c[n], e0 = wd0[n], e1 = wd1[n];
#pragma unroll
        for (int i = 0; i < 8; ++i) {
            int mm = mbase + i;
            float v = acc[nt][i] + bias + s_rad[w][mm] * c1 +
                      s_ea[w][mm][0] * e0 + s_ea[w][mm][1] * e1;
            v = silu_f(v);
            s_h[w][mm * HDIM + n] = (_Float16)v;
        }
    }

    // GEMM2: m2 = silu(m1 @ W2 + b2)
#pragma unroll
    for (int nt = 0; nt < 4; ++nt) acc[nt] = vzero;
    const v16h* B2 = (const v16h*)W2p;
#pragma unroll
    for (int kk = 0; kk < 2; ++kk) {
        v16h a = load_afrag(&s_h[w][0], HDIM, lane, kk * 32);
#pragma unroll
        for (int nt = 0; nt < 4; ++nt)
            acc[nt] = wmma_f16(a, B2[(kk * 4 + nt) * 32 + lane], acc[nt]);
    }
    // epilogue 2: scatter m into magg, restage for coord MLP
#pragma unroll
    for (int nt = 0; nt < 4; ++nt) {
        int n = nt * 16 + nl;
        float bias = b2[n];
#pragma unroll
        for (int i = 0; i < 8; ++i) {
            int mm = mbase + i;
            float v = silu_f(acc[nt][i] + bias);
            atomicAdd(&magg[(size_t)s_dst[w][mm] * HDIM + n], v);
            s_h[w][mm * HDIM + n] = (_Float16)v;
        }
    }

    // GEMM3: p = silu(m2 @ Cw1 + cb1); cm = p @ cw2 via LDS f32 atomics
#pragma unroll
    for (int nt = 0; nt < 4; ++nt) acc[nt] = vzero;
    const v16h* B3 = (const v16h*)Cw1p;
#pragma unroll
    for (int kk = 0; kk < 2; ++kk) {
        v16h a = load_afrag(&s_h[w][0], HDIM, lane, kk * 32);
#pragma unroll
        for (int nt = 0; nt < 4; ++nt)
            acc[nt] = wmma_f16(a, B3[(kk * 4 + nt) * 32 + lane], acc[nt]);
    }
#pragma unroll
    for (int nt = 0; nt < 4; ++nt) {
        int n = nt * 16 + nl;
        float bias = cb1[n], wv = cw2[n];
#pragma unroll
        for (int i = 0; i < 8; ++i) {
            int mm = mbase + i;
            float v = silu_f(acc[nt][i] + bias);
            atomicAdd(&s_cm[w][mm], v * wv);
        }
    }

    // trans = coord_diff * cm, scatter into agg
    if (lane < 16) {
        float cm = s_cm[w][lane];
        int d = s_dst[w][lane];
        atomicAdd(&agg[(size_t)d * 3 + 0], s_cd[w][lane][0] * cm);
        atomicAdd(&agg[(size_t)d * 3 + 1], s_cd[w][lane][1] * cm);
        atomicAdd(&agg[(size_t)d * 3 + 2], s_cd[w][lane][2] * cm);
    }
}

// ---------------------------------------------------------------------------
// Node kernel: one wave per 16-row tile over N*T rows.
// nin = [h | magg] (K=128) -> W1 -> silu -> W2 + b2, h += out; coord update.
// ---------------------------------------------------------------------------
__global__ __launch_bounds__(256) void node_kernel(
    _Float16* __restrict__ hh, float* __restrict__ h32, float* __restrict__ coord,
    const float* __restrict__ magg, const float* __restrict__ agg,
    const int* __restrict__ cnt,
    const _Float16* __restrict__ Nw1p, const _Float16* __restrict__ Nw2p,
    const float* __restrict__ nb1, const float* __restrict__ nb2, int numTiles) {
    __shared__ __align__(16) _Float16 s_n[8][16 * 128];

    int w = threadIdx.x >> 5;
    int lane = threadIdx.x & 31;
    int tile = blockIdx.x * 8 + w;
    if (tile >= numTiles) return;
    int tb = tile * 16;

    // stage nin = [h(f16) | magg(f32->f16)] rows of 128
    {
        int i = lane >> 1, part = lane & 1;
        int r = tb + i;
        const uint4* src = (const uint4*)(hh + (size_t)r * HDIM + part * 32);
        uint4* dst = (uint4*)(&s_n[w][i * 128 + part * 32]);
        dst[0] = src[0]; dst[1] = src[1]; dst[2] = src[2]; dst[3] = src[3];
        const float* ms = magg + (size_t)r * HDIM + part * 32;
        _Float16* md = &s_n[w][i * 128 + 64 + part * 32];
#pragma unroll
        for (int j = 0; j < 32; ++j) md[j] = (_Float16)ms[j];
    }

    const v8f vzero = {0.f, 0.f, 0.f, 0.f, 0.f, 0.f, 0.f, 0.f};
    v8f acc[4];
#pragma unroll
    for (int nt = 0; nt < 4; ++nt) acc[nt] = vzero;

    const v16h* B1 = (const v16h*)Nw1p;
#pragma unroll
    for (int kk = 0; kk < 4; ++kk) {
        v16h a = load_afrag(&s_n[w][0], 128, lane, kk * 32);
#pragma unroll
        for (int nt = 0; nt < 4; ++nt)
            acc[nt] = wmma_f16(a, B1[(kk * 4 + nt) * 32 + lane], acc[nt]);
    }

    int mbase = (lane >> 4) * 8;
    int nl = lane & 15;
#pragma unroll
    for (int nt = 0; nt < 4; ++nt) {
        int n = nt * 16 + nl;
        float bias = nb1[n];
#pragma unroll
        for (int i = 0; i < 8; ++i) {
            int mm = mbase + i;
            float v = silu_f(acc[nt][i] + bias);
            s_n[w][mm * HDIM + n] = (_Float16)v; // restage (stride 64)
        }
    }

#pragma unroll
    for (int nt = 0; nt < 4; ++nt) acc[nt] = vzero;
    const v16h* B2 = (const v16h*)Nw2p;
#pragma unroll
    for (int kk = 0; kk < 2; ++kk) {
        v16h a = load_afrag(&s_n[w][0], HDIM, lane, kk * 32);
#pragma unroll
        for (int nt = 0; nt < 4; ++nt)
            acc[nt] = wmma_f16(a, B2[(kk * 4 + nt) * 32 + lane], acc[nt]);
    }

    // residual writeback (f32 master + f16 mirror)
#pragma unroll
    for (int nt = 0; nt < 4; ++nt) {
        int n = nt * 16 + nl;
        float bias = nb2[n];
#pragma unroll
        for (int i = 0; i < 8; ++i) {
            int mm = mbase + i;
            size_t idx = (size_t)(tb + mm) * HDIM + n;
            float hv = h32[idx] + acc[nt][i] + bias;
            h32[idx] = hv;
            hh[idx] = (_Float16)hv;
        }
    }

    // coord += agg / max(cnt,1)
    if (lane < 16) {
        int r = tb + lane;
        int node = r / T_STEPS;
        int c = cnt[node];
        float cf = (float)(c > 0 ? c : 1);
#pragma unroll
        for (int k = 0; k < 3; ++k)
            coord[(size_t)r * 3 + k] += agg[(size_t)r * 3 + k] / cf;
    }
}

// x_hat[f][n][c] = softmax(theta[f]) . coord[n][:, c]
__global__ void out_kernel(const float* __restrict__ coord, const float* __restrict__ theta,
                           float* __restrict__ out, int Nn) {
    int idx = blockIdx.x * 256 + threadIdx.x;
    if (idx >= Nn * 3) return;
    int n = idx / 3;
    int c = idx % 3;
    float cv[T_STEPS];
#pragma unroll
    for (int t = 0; t < T_STEPS; ++t) cv[t] = coord[((size_t)n * T_STEPS + t) * 3 + c];
#pragma unroll
    for (int f = 0; f < FOUT; ++f) {
        float mx = -1e30f;
        for (int t = 0; t < T_STEPS; ++t) mx = fmaxf(mx, theta[f * T_STEPS + t]);
        float s = 0.f, a = 0.f;
        for (int t = 0; t < T_STEPS; ++t) {
            float e = __expf(theta[f * T_STEPS + t] - mx);
            s += e;
            a += e * cv[t];
        }
        out[((size_t)f * Nn + n) * 3 + c] = a / s;
    }
}

extern "C" void kernel_launch(void* const* d_in, const int* in_sizes, int n_in,
                              void* d_out, int out_size, void* d_ws, size_t ws_size,
                              hipStream_t stream) {
    (void)in_sizes; (void)n_in; (void)out_size; (void)ws_size;
    const float* h_in     = (const float*)d_in[0];
    const float* x_in     = (const float*)d_in[1];
    const int*   edges    = (const int*)d_in[2];
    const float* eattr    = (const float*)d_in[3];
    const float* emb_w    = (const float*)d_in[4];
    const float* emb_b    = (const float*)d_in[5];
    const float* time_emb = (const float*)d_in[6];
    const float* theta    = (const float*)d_in[7];
    const float* edge_w1  = (const float*)d_in[8];
    const float* edge_b1  = (const float*)d_in[9];
    const float* edge_w2  = (const float*)d_in[10];
    const float* edge_b2  = (const float*)d_in[11];
    const float* node_w1  = (const float*)d_in[12];
    const float* node_b1  = (const float*)d_in[13];
    const float* node_w2  = (const float*)d_in[14];
    const float* node_b2  = (const float*)d_in[15];
    const float* coord_w1 = (const float*)d_in[16];
    const float* coord_b1 = (const float*)d_in[17];
    const float* coord_w2 = (const float*)d_in[18];
    float* out = (float*)d_out;

    const int* rowIdx = edges;
    const int* colIdx = edges + NEDGES;

    char* ws = (char*)d_ws;
    size_t off = 0;
    auto alloc = [&](size_t bytes) -> size_t {
        size_t o = off;
        off += (bytes + 255) & ~(size_t)255;
        return o;
    };
    const size_t NT = (size_t)NNODES * T_STEPS;
    _Float16* hh    = (_Float16*)(ws + alloc(NT * HDIM * 2));
    float*    h32   = (float*)(ws + alloc(NT * HDIM * 4));
    float*    coord = (float*)(ws + alloc(NT * 3 * 4));
    float*    magg  = (float*)(ws + alloc(NT * HDIM * 4));
    float*    agg   = (float*)(ws + alloc(NT * 3 * 4));
    int*      cnt   = (int*)(ws + alloc((size_t)NNODES * 4));
    _Float16* packs = (_Float16*)(ws + alloc((size_t)NLAYERS * 28672 * 2));

    // pack all per-layer weights into WMMA B-fragment order (f16)
    for (int l = 0; l < NLAYERS; ++l) {
        _Float16* pl = packs + (size_t)l * 28672;
        const float* ew1 = edge_w1 + (size_t)l * 131 * 64;
        pack_kernel<<<16, 256, 0, stream>>>(ew1, pl, 64);                              // W1a
        pack_kernel<<<16, 256, 0, stream>>>(ew1 + 64 * 64, pl + 4096, 64);             // W1b
        pack_kernel<<<16, 256, 0, stream>>>(edge_w2 + (size_t)l * 4096, pl + 8192, 64);// W2
        pack_kernel<<<16, 256, 0, stream>>>(coord_w1 + (size_t)l * 4096, pl + 12288, 64); // Cw1
        pack_kernel<<<16, 256, 0, stream>>>(node_w2 + (size_t)l * 4096, pl + 16384, 64);  // Nw2
        pack_kernel<<<32, 256, 0, stream>>>(node_w1 + (size_t)l * 8192, pl + 20480, 128); // Nw1
    }

    // per-node edge counts (layer invariant)
    hipMemsetAsync(cnt, 0, (size_t)NNODES * 4, stream);
    count_kernel<<<(NEDGES + 255) / 256, 256, 0, stream>>>(rowIdx, cnt, NEDGES);

    // embedding + coordinate transpose
    {
        int total = (int)(NT * HDIM);
        embed_kernel<<<(total + 255) / 256, 256, 0, stream>>>(h_in, emb_w, emb_b,
                                                              time_emb, h32, hh, total);
        int tc = (int)(NT * 3);
        coordinit_kernel<<<(tc + 255) / 256, 256, 0, stream>>>(x_in, coord, tc);
    }

    const int edgeTiles = NEDGES * T_STEPS / 16; // 50000
    const int nodeTiles = (int)(NT / 16);        // 3125
    for (int l = 0; l < NLAYERS; ++l) {
        _Float16* pl = packs + (size_t)l * 28672;
        const float* ew1 = edge_w1 + (size_t)l * 131 * 64;
        hipMemsetAsync(magg, 0, NT * HDIM * 4, stream);
        hipMemsetAsync(agg, 0, NT * 3 * 4, stream);
        edge_kernel<<<(edgeTiles + 7) / 8, 256, 0, stream>>>(
            hh, coord, rowIdx, colIdx, eattr,
            pl, pl + 4096, pl + 8192, pl + 12288,
            edge_b1 + (size_t)l * 64, ew1 + 128 * 64, ew1 + 129 * 64, ew1 + 130 * 64,
            edge_b2 + (size_t)l * 64, coord_b1 + (size_t)l * 64, coord_w2 + (size_t)l * 64,
            magg, agg, edgeTiles);
        node_kernel<<<(nodeTiles + 7) / 8, 256, 0, stream>>>(
            hh, h32, coord, magg, agg, cnt,
            pl + 20480, pl + 16384, node_b1 + (size_t)l * 64, node_b2 + (size_t)l * 64,
            nodeTiles);
    }

    out_kernel<<<(NNODES * 3 + 255) / 256, 256, 0, stream>>>(coord, theta, out, NNODES);
}
